// Net_57604101374728
// MI455X (gfx1250) — matrix operand order repaired and checked by reading, compiled, not run on hardware
//
#include <hip/hip_runtime.h>
#include <hip/hip_bf16.h>

typedef __attribute__((ext_vector_type(16))) _Float16 v16h;
typedef __attribute__((ext_vector_type(8)))  float    v8f;

#define DEVI __device__ __forceinline__

// ---------------- BP group tables (q = 2,4,8,16,32,64 -> 126 channels) -------
__constant__ int GSTART[6] = {0, 2, 6, 14, 30, 62};
__constant__ int GSIZE[6]  = {2, 4, 8, 16, 32, 64};

DEVI int chan_group(int c) {
  if (c < 2) return 0; if (c < 6) return 1; if (c < 14) return 2;
  if (c < 30) return 3; if (c < 62) return 4; return 5;
}

DEVI unsigned hash_u32(unsigned x) {
  x ^= x >> 16; x *= 0x7feb352du; x ^= x >> 15; x *= 0x846ca68bu; x ^= x >> 16;
  return x;
}
DEVI float u01(unsigned h) { return (float)(h >> 8) * (1.0f / 16777216.0f) + 1e-7f; }

// ---------------- generic helpers -------------------------------------------
__global__ void zero_f32(float* __restrict__ p, long n) {
  long i = (long)blockIdx.x * blockDim.x + threadIdx.x;
  if (i < n) p[i] = 0.0f;
}

// ---------------- WMMA GEMM: C[M,N] = A*B (+bias)(+relu), batched ------------
// A element (m,k) = A[m*rsA + k*csA]; B element (k,n) = B[k*rsB + n*csB].
// One wave32 computes one 16x16 C tile; K in chunks of 32, f32->f16 convert on
// load, f32 accumulate via v_wmma_f32_16x16x32_f16.
// Branch-free operand fetch: out-of-range rows/cols are CLAMPED to a valid
// address and the loaded value is cndmask'ed to zero, so all 32 loads of a
// chunk issue as one clause with a single wait (no per-element exec branches).
// All in-tile index math is 32-bit; batch, A/B row-base and C-store offsets are 64-bit.
template <int EPI, int CSA>   // EPI: 0 none, 1 +bias, 2 +bias,relu. CSA: 1 => csA==1.
__global__ void wmma_gemm(const float* __restrict__ A, const float* __restrict__ B,
                          const float* __restrict__ bias, float* __restrict__ C,
                          int M, int N, int K,
                          int rsA, int csA, int rsB, int csB, int ldc,
                          long sA, long sB, long sC) {
  const int batch = blockIdx.y;
  A += (long)batch * sA; B += (long)batch * sB; C += (long)batch * sC;

  const int tilesN = (N + 15) / 16;
  const int tilesM = (M + 15) / 16;
  const int wave = (blockIdx.x * blockDim.x + threadIdx.x) >> 5;
  if (wave >= tilesM * tilesN) return;            // uniform per wave
  const int lane   = threadIdx.x & 31;
  const int laneHi = (lane >> 4) & 1;
  const int laneLo = lane & 15;
  const int tm = wave / tilesN, tn = wave % tilesN;
  const int row0 = tm * 16, col0 = tn * 16;

  const int icsA = (CSA == 1) ? 1 : csA;
  const int ar = row0 + laneLo;                   // A row this lane reads
  const int bc = col0 + laneLo;                   // B col this lane reads
  const bool aOk = (ar < M);
  const bool bOk = (bc < N);
  const float* __restrict__ Ap = A + (long)(aOk ? ar : 0) * rsA;
  const float* __restrict__ Bp = B + (long)(bOk ? bc : 0) * csB;
  const int aKofs = laneHi * 8;                   // fragment K base (A)
  const int bKofs = laneHi * 8;                   // fragment K base (B)

  union { v8f v; float e[8]; } acc = {};
  union { v16h v; _Float16 e[16]; } ua, ub;

  int k0 = 0;
  // ---- main chunks: K fully in range, only row/col masking needed ----------
  for (; k0 + 32 <= K; k0 += 32) {
    float af[16], bf[16];
    #pragma unroll
    for (int j = 0; j < 16; ++j) {                // issue all 32 loads, 1 wait
      const int ak = k0 + j + (j & 8) + aKofs;    // A: K = {0..7,16..23}+hi*8
      const int bk = k0 + j + (j & 8) + bKofs;    // B: K = {0..7,16..23}+hi*8
      af[j] = Ap[ak * icsA];
      bf[j] = Bp[bk * rsB];
    }
    #pragma unroll
    for (int j = 0; j < 16; ++j) {
      ua.e[j] = (_Float16)(aOk ? af[j] : 0.0f);
      ub.e[j] = (_Float16)(bOk ? bf[j] : 0.0f);
    }
    acc.v = __builtin_amdgcn_wmma_f32_16x16x32_f16(
        false, ua.v, false, ub.v, (short)0, acc.v, false, false);
  }
  // ---- tail chunk: clamp k too ---------------------------------------------
  if (k0 < K) {
    float af[16], bf[16];
    bool avOk[16], bvOk[16];
    #pragma unroll
    for (int j = 0; j < 16; ++j) {
      const int ak = k0 + j + (j & 8) + aKofs;
      const int bk = k0 + j + (j & 8) + bKofs;
      avOk[j] = aOk && (ak < K);
      bvOk[j] = bOk && (bk < K);
      af[j] = Ap[(avOk[j] ? ak : 0) * icsA];
      bf[j] = Bp[(bvOk[j] ? bk : 0) * rsB];
    }
    #pragma unroll
    for (int j = 0; j < 16; ++j) {
      ua.e[j] = (_Float16)(avOk[j] ? af[j] : 0.0f);
      ub.e[j] = (_Float16)(bvOk[j] ? bf[j] : 0.0f);
    }
    acc.v = __builtin_amdgcn_wmma_f32_16x16x32_f16(
        false, ua.v, false, ub.v, (short)0, acc.v, false, false);
  }

  #pragma unroll
  for (int r = 0; r < 8; ++r) {        // C: VGPR r -> M = r (+8 for hi lanes)
    const int row = row0 + r + laneHi * 8;
    const int col = col0 + laneLo;
    if (row < M && col < N) {
      float v = acc.e[r];
      if (EPI >= 1) v += bias[col];
      if (EPI == 2) v = fmaxf(v, 0.0f);
      C[(long)row * ldc + col] = v;
    }
  }
}

// ---------------- GCN (sparse) ----------------------------------------------
__global__ void deg_init(float* __restrict__ deg, int n) {
  int i = blockIdx.x * blockDim.x + threadIdx.x;
  if (i < n) deg[i] = 1.0f;                       // self loop
}
__global__ void deg_scatter(const int* __restrict__ dst, float* __restrict__ deg, int E) {
  int e = blockIdx.x * blockDim.x + threadIdx.x;
  if (e < E) atomicAdd(&deg[dst[e]], 1.0f);
}
__global__ void deg_finalize(float* __restrict__ deg, int n) {
  int i = blockIdx.x * blockDim.x + threadIdx.x;
  if (i < n) deg[i] = rsqrtf(deg[i]);             // deg >= 1 always
}
__global__ void gcn_self(const float* __restrict__ xw, const float* __restrict__ dinv,
                         const float* __restrict__ bias, float* __restrict__ y,
                         long n, int C) {
  long i = (long)blockIdx.x * blockDim.x + threadIdx.x;
  if (i >= n * C) return;
  int node = (int)(i / C), c = (int)(i % C);
  float dv = dinv[node];
  y[i] = xw[i] * dv * dv + bias[c];
}
__global__ void gcn_scatter(const float* __restrict__ xw, const int* __restrict__ src,
                            const int* __restrict__ dst, const float* __restrict__ dinv,
                            float* __restrict__ y, int E, int C) {
  long t = (long)blockIdx.x * blockDim.x + threadIdx.x;
  if (t >= (long)E * C) return;
  int e = (int)(t / C), c = (int)(t % C);
  int s = src[e], d = dst[e];
  atomicAdd(&y[(long)d * C + c], xw[(long)s * C + c] * dinv[s] * dinv[d]);
}

// ---------------- BatchNorm (training-mode batch stats) ---------------------
__global__ void bn_stats(const float* __restrict__ x, float* __restrict__ acc,
                         long n, int C) {
  __shared__ float ssum[32], ssq[32];
  if (threadIdx.x < (unsigned)C) { ssum[threadIdx.x] = 0.f; ssq[threadIdx.x] = 0.f; }
  __syncthreads();
  long total = n * C;
  for (long i = (long)blockIdx.x * blockDim.x + threadIdx.x; i < total;
       i += (long)gridDim.x * blockDim.x) {
    int c = (int)(i % C); float v = x[i];
    atomicAdd(&ssum[c], v); atomicAdd(&ssq[c], v * v);
  }
  __syncthreads();
  if (threadIdx.x < (unsigned)C) {
    atomicAdd(&acc[threadIdx.x], ssum[threadIdx.x]);
    atomicAdd(&acc[C + threadIdx.x], ssq[threadIdx.x]);
  }
}
__global__ void bn_apply(float* __restrict__ x, const float* __restrict__ acc,
                         const float* __restrict__ g, const float* __restrict__ be,
                         long n, int C) {
  long i = (long)blockIdx.x * blockDim.x + threadIdx.x;
  if (i >= n * C) return;
  int c = (int)(i % C);
  float m = acc[c] / (float)n;
  float v = acc[C + c] / (float)n - m * m;
  x[i] = g[c] * (x[i] - m) * rsqrtf(v + 1e-5f) + be[c];
}

// ---------------- Belief propagation (all 6 q-groups fused, width 126) ------
__global__ void bp_precompute(const float* __restrict__ betas, float* __restrict__ eb,
                              float* __restrict__ coef, int n) {
  int c = threadIdx.x;
  if (c >= 126) return;
  float b = betas[chan_group(c)];
  eb[c] = expm1f(b);
  coef[c] = 3.8f * b / (float)n;                  // MEAN_DEG * beta / n
}
__global__ void psi_init(float* __restrict__ psi, int n) {
  long i = (long)blockIdx.x * blockDim.x + threadIdx.x;
  if (i >= n) return;
  float* row = psi + i * 126;
  for (int g = 0; g < 6; ++g) {
    int s = GSTART[g], q = GSIZE[g];
    float sum = 0.f;
    for (int c = 0; c < q; ++c) {
      unsigned h1 = hash_u32(((unsigned)g * 0x9E3779B9u) ^
                             ((unsigned)i * 0x85EBCA6Bu) ^
                             ((unsigned)c * 0xC2B2AE35u) ^ 0x13579BDFu);
      unsigned h2 = hash_u32(h1 ^ 0x2468ACEFu);
      float z = sqrtf(-2.0f * __logf(u01(h1))) * __cosf(6.28318530718f * u01(h2));
      float e = __expf(z);                        // softmax(normal), |z| small
      row[s + c] = e; sum += e;
    }
    float inv = 1.0f / sum;
    for (int c = 0; c < q; ++c) row[s + c] *= inv;
  }
}
__global__ void bp_field(const float* __restrict__ psi, const float* __restrict__ coef,
                         float* __restrict__ field, int n) {
  int c = blockIdx.x;                              // 126 blocks
  __shared__ float s[256];
  float a = 0.f;
  for (int i = threadIdx.x; i < n; i += 256) a += psi[(long)i * 126 + c];
  s[threadIdx.x] = a; __syncthreads();
  for (int o = 128; o > 0; o >>= 1) {
    if (threadIdx.x < (unsigned)o) s[threadIdx.x] += s[threadIdx.x + o];
    __syncthreads();
  }
  if (threadIdx.x == 0) field[c] = coef[c] * s[0];
}
__global__ void bp_scatter(const float* __restrict__ psi, const int* __restrict__ src,
                           const int* __restrict__ dst, const float* __restrict__ eb,
                           float* __restrict__ msg, int E) {
  int e = blockIdx.x * blockDim.x + threadIdx.x;
  if (e >= E) return;
  long so = (long)src[e] * 126, dof = (long)dst[e] * 126;
  __builtin_prefetch(&psi[so], 0, 1);              // global_prefetch_b8
  const float2* __restrict__ ps = (const float2*)(psi + so);  // rows are 8B aligned
  #pragma unroll 3
  for (int c2 = 0; c2 < 63; ++c2) {                // 126 channels as 63 float2
    float2 p = ps[c2];
    atomicAdd(&msg[dof + 2 * c2 + 0], log1pf(eb[2 * c2 + 0] * p.x));
    atomicAdd(&msg[dof + 2 * c2 + 1], log1pf(eb[2 * c2 + 1] * p.y));
  }
}
__global__ void bp_softmax(float* __restrict__ msg, const float* __restrict__ field, int n) {
  long i = (long)blockIdx.x * blockDim.x + threadIdx.x;
  if (i >= n) return;
  float* row = msg + i * 126;
  for (int g = 0; g < 6; ++g) {
    int s = GSTART[g], q = GSIZE[g];
    float mx = -1e30f;
    for (int c = 0; c < q; ++c) mx = fmaxf(mx, row[s + c] - field[s + c]);
    float sum = 0.f;
    for (int c = 0; c < q; ++c) {
      float v = __expf(row[s + c] - field[s + c] - mx);
      row[s + c] = v; sum += v;
    }
    float inv = 1.0f / sum;
    for (int c = 0; c < q; ++c) row[s + c] *= inv;
  }
}

// ---------------- pooling-side kernels --------------------------------------
__global__ void softmax_row(float* __restrict__ x, long rows, int C) {
  long i = (long)blockIdx.x * blockDim.x + threadIdx.x;
  if (i >= rows) return;
  float* row = x + i * C;
  float mx = -1e30f;
  for (int c = 0; c < C; ++c) mx = fmaxf(mx, row[c]);
  float sum = 0.f;
  for (int c = 0; c < C; ++c) { float v = __expf(row[c] - mx); row[c] = v; sum += v; }
  float inv = 1.0f / sum;
  for (int c = 0; c < C; ++c) row[c] *= inv;
}
// T[src] += s1[dst]  (sparse A @ s1, duplicate edges sum naturally)
__global__ void t_scatter(const float* __restrict__ s1, const int* __restrict__ src,
                          const int* __restrict__ dst, float* __restrict__ T, int E) {
  long t = (long)blockIdx.x * blockDim.x + threadIdx.x;
  if (t >= (long)E * 100) return;
  int e = (int)(t / 100), l = (int)(t % 100);
  atomicAdd(&T[(long)src[e] * 100 + l], s1[(long)dst[e] * 100 + l]);
}
__global__ void adj_rowsum(const float* __restrict__ adj, float* __restrict__ d, int BI) {
  int i = blockIdx.x * blockDim.x + threadIdx.x;
  if (i >= BI) return;
  const float* row = adj + (long)i * 100;
  float s = 1.0f;                                  // + identity
  for (int j = 0; j < 100; ++j) s += row[j];
  d[i] = s;
}
__global__ void adj_norm(const float* __restrict__ adj, const float* __restrict__ d,
                         float* __restrict__ An, long total) {
  long t = (long)blockIdx.x * blockDim.x + threadIdx.x;
  if (t >= total) return;
  int b = (int)(t / 10000), r = (int)((t / 100) % 100), cl = (int)(t % 100);
  float a = adj[t] + (r == cl ? 1.0f : 0.0f);
  float di = d[b * 100 + r], dj = d[b * 100 + cl];
  float ri = di > 0.f ? rsqrtf(di) : 0.f;
  float rj = dj > 0.f ? rsqrtf(dj) : 0.f;
  An[t] = ri * a * rj;
}
__global__ void maxpool3(const float* __restrict__ a, const float* __restrict__ b,
                         const float* __restrict__ c, float* __restrict__ out,
                         int B, int N, int C) {
  int t = blockIdx.x * blockDim.x + threadIdx.x;
  if (t >= B * 3 * C) return;
  int bb = t / (3 * C), cc = t % (3 * C);
  const float* srcp = (cc < C) ? a : ((cc < 2 * C) ? b : c);
  int ch = cc % C;
  float mx = -1e30f;
  for (int node = 0; node < N; ++node)
    mx = fmaxf(mx, srcp[((long)bb * N + node) * C + ch]);
  out[t] = mx;
}
__global__ void concat2(const float* __restrict__ a, const float* __restrict__ b,
                        float* __restrict__ out, int B, int C) {
  int t = blockIdx.x * blockDim.x + threadIdx.x;
  if (t >= B * 2 * C) return;
  int bb = t / (2 * C), c = t % (2 * C);
  out[t] = (c < C) ? a[bb * C + c] : b[bb * C + (c - C)];
}
__global__ void tail_zero(float* __restrict__ out, int idx) {
  if (threadIdx.x == 0 && blockIdx.x == 0) out[idx] = 0.0f;
}

// ---------------- host side -------------------------------------------------
static void gemm(int epi, int aCont, const float* A, const float* B, const float* bias,
                 float* C, int M, int N, int K,
                 int rsA, int csA, int rsB, int csB, int ldc,
                 int batches, long sA, long sB, long sC, hipStream_t st) {
  int tiles = ((M + 15) / 16) * ((N + 15) / 16);
  dim3 grid((tiles + 7) / 8, batches), block(256);
  if (aCont) {
    if (epi == 0)      wmma_gemm<0, 1><<<grid, block, 0, st>>>(A, B, bias, C, M, N, K, rsA, csA, rsB, csB, ldc, sA, sB, sC);
    else if (epi == 1) wmma_gemm<1, 1><<<grid, block, 0, st>>>(A, B, bias, C, M, N, K, rsA, csA, rsB, csB, ldc, sA, sB, sC);
    else               wmma_gemm<2, 1><<<grid, block, 0, st>>>(A, B, bias, C, M, N, K, rsA, csA, rsB, csB, ldc, sA, sB, sC);
  } else {
    if (epi == 0)      wmma_gemm<0, 0><<<grid, block, 0, st>>>(A, B, bias, C, M, N, K, rsA, csA, rsB, csB, ldc, sA, sB, sC);
    else if (epi == 1) wmma_gemm<1, 0><<<grid, block, 0, st>>>(A, B, bias, C, M, N, K, rsA, csA, rsB, csB, ldc, sA, sB, sC);
    else               wmma_gemm<2, 0><<<grid, block, 0, st>>>(A, B, bias, C, M, N, K, rsA, csA, rsB, csB, ldc, sA, sB, sC);
  }
}

extern "C" void kernel_launch(void* const* d_in, const int* in_sizes, int n_in,
                              void* d_out, int out_size, void* d_ws, size_t ws_size,
                              hipStream_t stream) {
  const int n = in_sizes[0] / 3;        // 65536
  const int E = in_sizes[1] / 2;        // 524288
  const int B = 64;                     // num_graphs (setup constant)
  const int N = n / B;                  // 1024

  const float* x    = (const float*)d_in[0];
  const int*   src  = (const int*)d_in[1];
  const int*   dst  = src + E;
  const float* W1[3]  = {(const float*)d_in[3],  (const float*)d_in[7],  (const float*)d_in[11]};
  const float* Bb1[3] = {(const float*)d_in[4],  (const float*)d_in[8],  (const float*)d_in[12]};
  const float* G1[3]  = {(const float*)d_in[5],  (const float*)d_in[9],  (const float*)d_in[13]};
  const float* Be1[3] = {(const float*)d_in[6],  (const float*)d_in[10], (const float*)d_in[14]};
  const float* W2[3]  = {(const float*)d_in[15], (const float*)d_in[19], (const float*)d_in[23]};
  const float* Bb2[3] = {(const float*)d_in[16], (const float*)d_in[20], (const float*)d_in[24]};
  const float* G2[3]  = {(const float*)d_in[17], (const float*)d_in[21], (const float*)d_in[25]};
  const float* Be2[3] = {(const float*)d_in[18], (const float*)d_in[22], (const float*)d_in[26]};
  const float* betas  = (const float*)d_in[27];
  const float* pool_w = (const float*)d_in[28];
  const float* pool_b = (const float*)d_in[29];
  const float* fc1_w  = (const float*)d_in[30];
  const float* fc1_b  = (const float*)d_in[31];
  const float* fc2_w  = (const float*)d_in[32];
  const float* fc2_b  = (const float*)d_in[33];
  float* dout = (float*)d_out;

  // ---- bump allocator over d_ws ----
  size_t off = 0;
  auto alloc = [&](size_t nf) -> float* {
    float* p = (float*)((char*)d_ws + off);
    off += ((nf * sizeof(float)) + 255) & ~(size_t)255;
    return p;
  };
  float* dinv  = alloc(n);
  float* xw    = alloc((size_t)n * 30);
  float* x11   = alloc((size_t)n * 30);
  float* x12   = alloc((size_t)n * 30);
  float* x13   = alloc((size_t)n * 30);
  float* bnacc = alloc(64);
  float* psiA  = alloc((size_t)n * 126);
  float* psiB  = alloc((size_t)n * 126);   // doubles as msg buffer / s1
  float* eb    = alloc(128);
  float* coef  = alloc(128);
  float* field = alloc(128);
  float* T     = alloc((size_t)n * 100);
  float* p1x   = alloc((size_t)B * 100 * 30);
  float* p1adj = alloc((size_t)B * 100 * 100);
  float* An    = alloc((size_t)B * 100 * 100);
  float* drow  = alloc((size_t)B * 100);
  float* xw2   = alloc((size_t)B * 100 * 30);
  float* x21   = alloc((size_t)B * 100 * 30);
  float* x22   = alloc((size_t)B * 100 * 30);
  float* x23   = alloc((size_t)B * 100 * 30);
  float* x1out = alloc((size_t)B * 90);
  float* x2out = alloc((size_t)B * 90);
  float* convo = alloc((size_t)B * 180);
  float* fc1o  = alloc((size_t)B * 50);
  (void)ws_size; (void)n_in;

  auto cdiv = [](long a, long b) -> unsigned { return (unsigned)((a + b - 1) / b); };
  auto zero = [&](float* p, long cnt) {
    zero_f32<<<dim3(cdiv(cnt, 256)), dim3(256), 0, stream>>>(p, cnt);
  };

  // ---- degrees (deg over dst + self loop), dinv = deg^-1/2 ----
  deg_init<<<cdiv(n, 256), 256, 0, stream>>>(dinv, n);
  deg_scatter<<<cdiv(E, 256), 256, 0, stream>>>(dst, dinv, E);
  deg_finalize<<<cdiv(n, 256), 256, 0, stream>>>(dinv, n);

  // ---- stage 1: three sparse GCN + BN layers ----
  {
    const float* xin = x; int Kin = 3;
    float* outs[3] = {x11, x12, x13};
    for (int l = 0; l < 3; ++l) {
      gemm(0, 1, xin, W1[l], nullptr, xw, n, 30, Kin, Kin, 1, 30, 1, 30, 1, 0, 0, 0, stream);
      gcn_self<<<cdiv((long)n * 30, 256), 256, 0, stream>>>(xw, dinv, Bb1[l], outs[l], n, 30);
      gcn_scatter<<<cdiv((long)E * 30, 256), 256, 0, stream>>>(xw, src, dst, dinv, outs[l], E, 30);
      zero(bnacc, 64);
      bn_stats<<<512, 256, 0, stream>>>(outs[l], bnacc, n, 30);
      bn_apply<<<cdiv((long)n * 30, 256), 256, 0, stream>>>(outs[l], bnacc, G1[l], Be1[l], n, 30);
      xin = outs[l]; Kin = 30;
    }
  }

  // ---- belief propagation: fused 126-channel sweeps, 10 iterations ----
  bp_precompute<<<1, 128, 0, stream>>>(betas, eb, coef, n);
  psi_init<<<cdiv(n, 256), 256, 0, stream>>>(psiA, n);
  {
    float* cur = psiA; float* oth = psiB;
    for (int it = 0; it < 10; ++it) {
      bp_field<<<126, 256, 0, stream>>>(cur, coef, field, n);
      zero(oth, (long)n * 126);
      bp_scatter<<<cdiv(E, 256), 256, 0, stream>>>(cur, src, dst, eb, oth, E);
      bp_softmax<<<cdiv(n, 256), 256, 0, stream>>>(oth, field, n);
      float* t = cur; cur = oth; oth = t;
    }
    // 10 iterations -> cur == psiA
  }
  float* s1 = psiB;   // reuse msg buffer (n*100 <= n*126)

  // ---- pooling assignment: s1 = softmax(psi @ pool_w + pool_b) ----
  gemm(1, 1, psiA, pool_w, pool_b, s1, n, 100, 126, 126, 1, 100, 1, 100, 1, 0, 0, 0, stream);
  softmax_row<<<cdiv(n, 256), 256, 0, stream>>>(s1, n, 100);

  // ---- sparse A @ s1 -> T, then WMMA batched s1^T @ {x13, T} ----
  zero(T, (long)n * 100);
  t_scatter<<<cdiv((long)E * 100, 256), 256, 0, stream>>>(s1, src, dst, T, E);
  // p1_x[b] = s1[b]^T (100x1024) @ x13[b] (1024x30)
  gemm(0, 0, s1, x13, nullptr, p1x, 100, 30, N, 1, 100, 30, 1, 30,
       B, (long)N * 100, (long)N * 30, 3000, stream);
  // p1_adj[b] = s1[b]^T @ T[b]  (100x100)
  gemm(0, 0, s1, T, nullptr, p1adj, 100, 100, N, 1, 100, 100, 1, 100,
       B, (long)N * 100, (long)N * 100, 10000, stream);

  // ---- normalized pooled adjacency ----
  adj_rowsum<<<cdiv(B * 100, 256), 256, 0, stream>>>(p1adj, drow, B * 100);
  adj_norm<<<cdiv((long)B * 10000, 256), 256, 0, stream>>>(p1adj, drow, An, (long)B * 10000);

  // ---- stage 2: three dense GCN + BN layers on pooled graphs ----
  {
    const float* xin2 = p1x;
    float* outs2[3] = {x21, x22, x23};
    for (int l = 0; l < 3; ++l) {
      gemm(0, 1, xin2, W2[l], nullptr, xw2, B * 100, 30, 30, 30, 1, 30, 1, 30, 1, 0, 0, 0, stream);
      gemm(1, 1, An, xw2, Bb2[l], outs2[l], 100, 30, 100, 100, 1, 30, 1, 30,
           B, 10000, 3000, 3000, stream);
      zero(bnacc, 64);
      bn_stats<<<256, 256, 0, stream>>>(outs2[l], bnacc, (long)B * 100, 30);
      bn_apply<<<cdiv((long)B * 3000, 256), 256, 0, stream>>>(outs2[l], bnacc, G2[l], Be2[l],
                                                              (long)B * 100, 30);
      xin2 = outs2[l];
    }
  }

  // ---- readout: max pools, concat, FC head ----
  maxpool3<<<cdiv(B * 90, 256), 256, 0, stream>>>(x11, x12, x13, x1out, B, N, 30);
  maxpool3<<<cdiv(B * 90, 256), 256, 0, stream>>>(x21, x22, x23, x2out, B, 100, 30);
  concat2<<<cdiv(B * 180, 256), 256, 0, stream>>>(x1out, x2out, convo, B, 90);
  gemm(2, 1, convo, fc1_w, fc1_b, fc1o, B, 50, 180, 180, 1, 50, 1, 50, 1, 0, 0, 0, stream);
  gemm(1, 1, fc1o, fc2_w, fc2_b, dout, B, 6, 50, 50, 1, 6, 1, 6, 1, 0, 0, 0, stream);
  tail_zero<<<1, 32, 0, stream>>>(dout, out_size - 1);   // aux zeros((1,))
}
